// BiaffineSpan2WordLabeler_48576080118495
// MI455X (gfx1250) — matrix operand order repaired and checked by reading, not compiled, not run on hardware
//
#include <hip/hip_runtime.h>
#include <cstddef>

typedef __attribute__((ext_vector_type(16))) __bf16 v16bf;
typedef __attribute__((ext_vector_type(8)))  __bf16 v8bf;
typedef __attribute__((ext_vector_type(8)))  float  v8f;
typedef __attribute__((ext_vector_type(4)))  float  v4f;

#define SCALE_F 0.21022410381342865f   // 1 / 512^0.25

// ---------- fragment loaders (CDNA5 16-bit WMMA VGPR layouts) ----------

// B 32x16 bf16: lane holds K chunks +0..7 and +16..23 of column n=lane&15; upper half-wave K+8.
__device__ __forceinline__ v16bf load_bfrag(const __bf16* colptr /*already + ((lane&16)>>1)*/) {
  const v8bf* p = (const v8bf*)colptr;
  v8bf lo = p[0], hi = p[2];
  v16bf r;
#pragma unroll
  for (int e = 0; e < 8; ++e) { r[e] = lo[e]; r[e + 8] = hi[e]; }
  return r;
}

// A 16x32 bf16 from a bf16 row: chunks at K = koff and koff+16
__device__ __forceinline__ v16bf load_afrag_bf(const __bf16* rowptr, int kbase, int koff) {
  const v8bf* p0 = (const v8bf*)(rowptr + kbase + koff);
  const v8bf* p1 = (const v8bf*)(rowptr + kbase + koff + 16);
  v8bf lo = p0[0], hi = p1[0];
  v16bf r;
#pragma unroll
  for (int e = 0; e < 8; ++e) { r[e] = lo[e]; r[e + 8] = hi[e]; }
  return r;
}

// A 16x32 bf16 from an f32 row (convert on the fly)
__device__ __forceinline__ v16bf cvt_afrag_f32(const float* rowptr, int kbase, int koff) {
  const v4f* p = (const v4f*)(rowptr + kbase + koff);
  v4f d0 = p[0], d1 = p[1], d2 = p[4], d3 = p[5];   // +0..7 and +16..23
  v16bf r;
#pragma unroll
  for (int e = 0; e < 4; ++e) {
    r[e]      = (__bf16)d0[e];
    r[e + 4]  = (__bf16)d1[e];
    r[e + 8]  = (__bf16)d2[e];
    r[e + 12] = (__bf16)d3[e];
  }
  return r;
}

// Two A 16x32 bf16 fragments as diffs of two j-rows against one shared i-row.
__device__ __forceinline__ void afrag_diff2(const float* jp0, const float* jp1,
                                            const float* ip, int kbase, int koff,
                                            v16bf& a0, v16bf& a1) {
  const v4f* pi = (const v4f*)(ip + kbase + koff);
  v4f i0 = pi[0], i1 = pi[1], i2 = pi[4], i3 = pi[5];
  const v4f* p0 = (const v4f*)(jp0 + kbase + koff);
  v4f x0 = p0[0] - i0, x1 = p0[1] - i1, x2 = p0[4] - i2, x3 = p0[5] - i3;
  const v4f* p1 = (const v4f*)(jp1 + kbase + koff);
  v4f y0 = p1[0] - i0, y1 = p1[1] - i1, y2 = p1[4] - i2, y3 = p1[5] - i3;
#pragma unroll
  for (int e = 0; e < 4; ++e) {
    a0[e] = (__bf16)x0[e]; a0[e + 4] = (__bf16)x1[e];
    a0[e + 8] = (__bf16)x2[e]; a0[e + 12] = (__bf16)x3[e];
    a1[e] = (__bf16)y0[e]; a1[e + 4] = (__bf16)y1[e];
    a1[e + 8] = (__bf16)y2[e]; a1[e + 12] = (__bf16)y3[e];
  }
}

// B 32x16 bf16 from f32 at +0..7 and +16..23 (convert on the fly)
__device__ __forceinline__ v16bf cvt_bfrag_f32(const float* colptr) {
  const v4f* p = (const v4f*)colptr;
  v4f c0 = p[0], c1 = p[1], c2 = p[4], c3 = p[5];
  v16bf r;
#pragma unroll
  for (int e = 0; e < 4; ++e) {
    r[e]      = (__bf16)c0[e];
    r[e + 4]  = (__bf16)c1[e];
    r[e + 8]  = (__bf16)c2[e];
    r[e + 12] = (__bf16)c3[e];
  }
  return r;
}

#define WMMA_BF16(a, b, c) \
  __builtin_amdgcn_wmma_f32_16x16x32_bf16(false, (a), false, (b), (short)0, (c), false, false)

// ---------- kernel 0: transpose + f32->bf16 weights (K x N -> N x K) ----------
__global__ __launch_bounds__(256) void k_cvt_tr(const float* __restrict__ in,
                                               __bf16* __restrict__ out) {
  const int K = 1024, N = 512;
  int idx = blockIdx.x * blockDim.x + threadIdx.x;
  if (idx >= K * N) return;
  int n = idx >> 10;        // idx / K
  int k = idx & 1023;       // idx % K
  out[(size_t)n * K + k] = (__bf16)in[(size_t)k * N + n];
}

// ---------- kernel 1: h2 = leaky(x_dep @ W2 + b2) * SCALE  (f32 out) ----------
__global__ __launch_bounds__(256) void k_h2(const float* __restrict__ xd,
                                            const __bf16* __restrict__ W2t,
                                            const float* __restrict__ b2,
                                            float* __restrict__ h2) {
  int wave = blockIdx.x * 8 + (threadIdx.x >> 5);
  int lane = threadIdx.x & 31;
  int nt = wave & 7, mt = wave >> 3;                 // 8 n-tiles x 32 m-tiles
  int m0 = mt * 16;
  int koff = (lane & 16) >> 1;                       // 0 or 8
  int bhalf = (lane & 16) >> 1;                      // 0 or 8 (B K-offset)
  int nlan = lane & 15;
  const float* arow = xd + (size_t)(m0 + (lane & 15)) * 1024;
  v8f acc[4] = {};
  for (int kk = 0; kk < 1024; kk += 32) {
    v16bf a = cvt_afrag_f32(arow, kk, koff);
#pragma unroll
    for (int t = 0; t < 4; ++t) {
      int n = nt * 64 + t * 16 + nlan;
      v16bf b = load_bfrag(W2t + (size_t)n * 1024 + kk + bhalf);
      acc[t] = WMMA_BF16(a, b, acc[t]);
    }
  }
  int mb = m0 + ((lane & 16) >> 1);                  // +8 rows for upper half-wave
#pragma unroll
  for (int t = 0; t < 4; ++t) {
    int n = nt * 64 + t * 16 + nlan;
    float bias = b2[n];
#pragma unroll
    for (int r = 0; r < 8; ++r) {
      float v = acc[t][r] + bias;
      v = (v >= 0.f) ? v : 0.1f * v;
      h2[(size_t)(mb + r) * 512 + n] = v * SCALE_F;
    }
  }
}

// ---------- kernel 2: t[bo,y,i] = sum_j Wa[o,i,j] * h2[b,y,j]  (bf16, stride 528) ----------
__global__ __launch_bounds__(256) void k_t(const float* __restrict__ h2,
                                           const float* __restrict__ Wa,
                                           __bf16* __restrict__ tb) {
  int wave = blockIdx.x * 8 + (threadIdx.x >> 5);
  int lane = threadIdx.x & 31;
  int bo = wave >> 6, loc = wave & 63;
  int mt = loc >> 3, nt = loc & 7;                   // y-tile, i-tile
  int b = bo >> 2, o = bo & 3;
  int y0 = mt * 16;
  int koff = (lane & 16) >> 1;
  int bhalf = (lane & 16) >> 1;
  int nlan = lane & 15;
  const float* arow = h2 + (size_t)(b * 128 + y0 + (lane & 15)) * 512;
  v8f acc[4] = {};
  for (int kk = 0; kk < 512; kk += 32) {
    v16bf a = cvt_afrag_f32(arow, kk, koff);
#pragma unroll
    for (int t = 0; t < 4; ++t) {
      int i = nt * 64 + t * 16 + nlan;
      // Wa[o][i][j]: j contiguous == K contiguous -> direct B fragment
      v16bf bf = cvt_bfrag_f32(Wa + ((size_t)o * 513 + i) * 512 + kk + bhalf);
      acc[t] = WMMA_BF16(a, bf, acc[t]);
    }
  }
  int yb = y0 + ((lane & 16) >> 1);
#pragma unroll
  for (int t = 0; t < 4; ++t) {
    int i = nt * 64 + t * 16 + nlan;
#pragma unroll
    for (int r = 0; r < 8; ++r)
      tb[(size_t)(bo * 128 + yb + r) * 528 + i] = (__bf16)acc[t][r];
  }
}

// ---------- kernel 2b: bias column t[bo,y,512] = dot(Wa[o,512,:], h2[b,y,:]) ----------
__global__ __launch_bounds__(256) void k_tbias(const float* __restrict__ h2,
                                               const float* __restrict__ Wa,
                                               __bf16* __restrict__ tb) {
  int idx = blockIdx.x * blockDim.x + threadIdx.x;
  if (idx >= 4 * 4 * 128) return;
  int y = idx & 127, bo = idx >> 7;
  int b = bo >> 2, o = bo & 3;
  const float* hr = h2 + (size_t)(b * 128 + y) * 512;
  const float* wr = Wa + ((size_t)o * 513 + 512) * 512;
  float s = 0.f;
  for (int k = 0; k < 512; ++k) s += hr[k] * wr[k];
  tb[(size_t)(bo * 128 + y) * 528 + 512] = (__bf16)s;
}

// ---------- kernel 3: h1 = leaky(diff @ W1 + b1) * SCALE  (bf16 out) ----------
// 32x64 tile per wave, software-pipelined (double-buffered fragments).
__global__ __launch_bounds__(256) void k_h1(const float* __restrict__ xc,
                                            const __bf16* __restrict__ W1t,
                                            const float* __restrict__ b1,
                                            __bf16* __restrict__ h1) {
  int wave = blockIdx.x * 8 + (threadIdx.x >> 5);
  int lane = threadIdx.x & 31;
  int nt = wave & 7, mt = wave >> 3;                 // 8 n-tiles x 2048 m-tiles
  int m0 = mt * 32;
  int b = m0 >> 14;                                  // / 16384
  int rem = m0 & 16383;
  int irow = rem >> 7;                               // span start i (uniform in tile)
  int j0 = rem & 127;                                // multiple of 32
  const float* xb = xc + (size_t)b * 128 * 1024;
  const float* jp0 = xb + (size_t)(j0 + (lane & 15)) * 1024;
  const float* jp1 = jp0 + 16 * 1024;
  const float* ip  = xb + (size_t)irow * 1024;
  int koff = (lane & 16) >> 1;
  int bhalf = (lane & 16) >> 1;
  int nlan = lane & 15;
  const __bf16* wp[4];
#pragma unroll
  for (int t = 0; t < 4; ++t)
    wp[t] = W1t + (size_t)(nt * 64 + t * 16 + nlan) * 1024 + bhalf;

  v8f acc[8] = {};
  v16bf a0, a1, bf[4];
  afrag_diff2(jp0, jp1, ip, 0, koff, a0, a1);
#pragma unroll
  for (int t = 0; t < 4; ++t) bf[t] = load_bfrag(wp[t]);

#pragma unroll 2
  for (int kk = 32; kk < 1024; kk += 32) {
    v16bf na0, na1, nb[4];
    afrag_diff2(jp0, jp1, ip, kk, koff, na0, na1);   // prefetch next K-step
#pragma unroll
    for (int t = 0; t < 4; ++t) nb[t] = load_bfrag(wp[t] + kk);
#pragma unroll
    for (int t = 0; t < 4; ++t) {
      acc[t]     = WMMA_BF16(a0, bf[t], acc[t]);
      acc[t + 4] = WMMA_BF16(a1, bf[t], acc[t + 4]);
    }
    a0 = na0; a1 = na1;
#pragma unroll
    for (int t = 0; t < 4; ++t) bf[t] = nb[t];
  }
#pragma unroll
  for (int t = 0; t < 4; ++t) {                      // drain
    acc[t]     = WMMA_BF16(a0, bf[t], acc[t]);
    acc[t + 4] = WMMA_BF16(a1, bf[t], acc[t + 4]);
  }

  int mloc = (lane & 16) >> 1;
#pragma unroll
  for (int h = 0; h < 2; ++h) {
    int mb = m0 + h * 16 + mloc;
#pragma unroll
    for (int t = 0; t < 4; ++t) {
      int n = nt * 64 + t * 16 + nlan;
      float bias = b1[n];
#pragma unroll
      for (int r = 0; r < 8; ++r) {
        float v = acc[h * 4 + t][r] + bias;
        v = (v >= 0.f) ? v : 0.1f * v;
        h1[(size_t)(mb + r) * 512 + n] = (__bf16)(v * SCALE_F);
      }
    }
  }
}

// ---------- kernel 4: out[bo,x,y] = h1[bx,:] . t[bo,y,:] + t[bo,y,512] ----------
// 32x64 tile per wave, software-pipelined.
__global__ __launch_bounds__(256) void k_out(const __bf16* __restrict__ h1,
                                             const __bf16* __restrict__ tb,
                                             float* __restrict__ out) {
  int wave = blockIdx.x * 8 + (threadIdx.x >> 5);
  int lane = threadIdx.x & 31;
  int bo = wave >> 10, loc = wave & 1023;
  int mt = loc >> 1, nt = loc & 1;                   // 512 x-tiles, 2 y-tiles
  int b = bo >> 2;
  int x0 = mt * 32;
  int koff = (lane & 16) >> 1;
  int bhalf = (lane & 16) >> 1;
  int nlan = lane & 15;
  const __bf16* ar0 = h1 + (size_t)(b * 16384 + x0 + (lane & 15)) * 512;
  const __bf16* ar1 = ar0 + (size_t)16 * 512;
  const __bf16* tp[4];
#pragma unroll
  for (int t = 0; t < 4; ++t)
    tp[t] = tb + (size_t)(bo * 128 + nt * 64 + t * 16 + nlan) * 528 + bhalf;

  v8f acc[8] = {};
  v16bf a0, a1, bf[4];
  a0 = load_afrag_bf(ar0, 0, koff);
  a1 = load_afrag_bf(ar1, 0, koff);
#pragma unroll
  for (int t = 0; t < 4; ++t) bf[t] = load_bfrag(tp[t]);

#pragma unroll 2
  for (int kk = 32; kk < 512; kk += 32) {
    v16bf na0 = load_afrag_bf(ar0, kk, koff);
    v16bf na1 = load_afrag_bf(ar1, kk, koff);
    v16bf nb[4];
#pragma unroll
    for (int t = 0; t < 4; ++t) nb[t] = load_bfrag(tp[t] + kk);
#pragma unroll
    for (int t = 0; t < 4; ++t) {
      acc[t]     = WMMA_BF16(a0, bf[t], acc[t]);
      acc[t + 4] = WMMA_BF16(a1, bf[t], acc[t + 4]);
    }
    a0 = na0; a1 = na1;
#pragma unroll
    for (int t = 0; t < 4; ++t) bf[t] = nb[t];
  }
#pragma unroll
  for (int t = 0; t < 4; ++t) {                      // drain
    acc[t]     = WMMA_BF16(a0, bf[t], acc[t]);
    acc[t + 4] = WMMA_BF16(a1, bf[t], acc[t + 4]);
  }

  int mloc = (lane & 16) >> 1;
#pragma unroll
  for (int t = 0; t < 4; ++t) {
    int y = nt * 64 + t * 16 + nlan;
    float bias = (float)tb[(size_t)(bo * 128 + y) * 528 + 512];  // h1b ones column
#pragma unroll
    for (int h = 0; h < 2; ++h) {
      int xb = x0 + h * 16 + mloc;
#pragma unroll
      for (int r = 0; r < 8; ++r)
        out[((size_t)bo * 16384 + xb + r) * 128 + y] = acc[h * 4 + t][r] + bias;
    }
  }
}

extern "C" void kernel_launch(void* const* d_in, const int* in_sizes, int n_in,
                              void* d_out, int out_size, void* d_ws, size_t ws_size,
                              hipStream_t stream) {
  (void)in_sizes; (void)n_in; (void)out_size; (void)ws_size;
  const float* x_const = (const float*)d_in[0];
  const float* x_dep   = (const float*)d_in[1];
  const float* W1      = (const float*)d_in[2];
  const float* b1      = (const float*)d_in[3];
  const float* W2      = (const float*)d_in[4];
  const float* b2      = (const float*)d_in[5];
  const float* Wa      = (const float*)d_in[6];
  float* out = (float*)d_out;

  char* ws = (char*)d_ws;
  __bf16* W1t = (__bf16*)(ws);                        // 1 MB  (512 x 1024 bf16)
  __bf16* W2t = (__bf16*)(ws + (1u << 20));           // 1 MB
  float*  h2  = (float*) (ws + (2u << 20));           // 1 MB  (512 x 512 f32)
  __bf16* tb  = (__bf16*)(ws + (3u << 20));           // ~2.1 MB (16 x 128 x 528 bf16)
  __bf16* h1  = (__bf16*)(ws + (6u << 20));           // 64 MB (65536 x 512 bf16)

  k_cvt_tr<<<2048, 256, 0, stream>>>(W1, W1t);
  k_cvt_tr<<<2048, 256, 0, stream>>>(W2, W2t);
  k_h2   <<<32,   256, 0, stream>>>(x_dep, W2t, b2, h2);
  k_t    <<<128,  256, 0, stream>>>(h2, Wa, tb);
  k_tbias<<<8,    256, 0, stream>>>(h2, Wa, tb);
  k_h1   <<<2048, 256, 0, stream>>>(x_const, W1t, b1, h1);
  k_out  <<<2048, 256, 0, stream>>>(h1, tb, out);
}
